// MHSA_TDC_74380243632642
// MI455X (gfx1250) — matrix operand order repaired and checked by reading, not compiled, not run on hardware
//
#include <hip/hip_runtime.h>
#include <hip/hip_bf16.h>

typedef __attribute__((ext_vector_type(16))) _Float16 v16h;
typedef __attribute__((ext_vector_type(8)))  _Float16 v8h;
typedef __attribute__((ext_vector_type(8)))  float    v8f;

#define NB   2
#define NT   64
#define NHs  8
#define NWs  8
#define NP   4096          // NT*NHs*NWs
#define NC   128
#define NHEAD 4
#define ND   32
#define THETA 0.7f

// ---------------------------------------------------------------------------
// 1) convert x (f32, [B,P,C] channel-last == [B,T,H,W,C]) to f16
// ---------------------------------------------------------------------------
__global__ __launch_bounds__(256)
void xcvt_kernel(const float* __restrict__ x, _Float16* __restrict__ xh, int n) {
  int i = blockIdx.x * 256 + threadIdx.x;
  if (i < n) xh[i] = (_Float16)x[i];
}

// ---------------------------------------------------------------------------
// 2) repack weights:
//    Wq/Wk [O,I,3,3,3] -> f16 [tap][ci][co], center tap (13) -= theta*kd
//    Wv [O,I,1,1,1]    -> f16 [ci][co]
// ---------------------------------------------------------------------------
__global__ __launch_bounds__(256)
void weight_prep_kernel(const float* __restrict__ Wq, const float* __restrict__ Wk,
                        const float* __restrict__ Wv,
                        _Float16* __restrict__ WqT, _Float16* __restrict__ WkT,
                        _Float16* __restrict__ WvT) {
  int idx = blockIdx.x * 256 + threadIdx.x;
  if (idx >= NC * NC) return;
  int co = idx >> 7, ci = idx & 127;
  if (blockIdx.y == 2) {
    WvT[(size_t)ci * NC + co] = (_Float16)Wv[(size_t)co * NC + ci];
    return;
  }
  const float* W  = (blockIdx.y == 0) ? Wq  : Wk;
  _Float16*    WT = (blockIdx.y == 0) ? WqT : WkT;
  const float* wbase = W + ((size_t)co * NC + ci) * 27;
  float kd = 0.f;
  #pragma unroll
  for (int s = 0; s < 9; ++s) kd += wbase[s] + wbase[18 + s];   // temporal slices 0, 2
  #pragma unroll
  for (int tap = 0; tap < 27; ++tap) {
    float wv = wbase[tap];
    if (tap == 13) wv -= THETA * kd;                            // fold -theta*out_diff
    WT[((size_t)tap * NC + ci) * NC + co] = (_Float16)wv;
  }
}

// ---------------------------------------------------------------------------
// helper: A-fragment (16x32 f16) for one row per lane, row base ptr + k-chunk
// ---------------------------------------------------------------------------
__device__ __forceinline__ v16h loadAfrag(const _Float16* xrow, bool valid, int kc, int ks) {
  v16h a;
  if (valid) {
    const _Float16* pa = xrow + kc * 32 + ks;
    v8h lo  = *(const v8h*)(pa);
    v8h hiv = *(const v8h*)(pa + 16);
    #pragma unroll
    for (int j = 0; j < 8; ++j) { a[j] = lo[j]; a[8 + j] = hiv[j]; }
  } else {
    #pragma unroll
    for (int j = 0; j < 16; ++j) a[j] = (_Float16)0.0f;
  }
  return a;
}

// ---------------------------------------------------------------------------
// 3) Q/K CDC-conv as implicit GEMM, 2x2 WMMA register blocking.
//    One wave -> 32x32 output tile. grid.z: bit0 = batch, bit1 = q/k select.
// ---------------------------------------------------------------------------
__global__ __launch_bounds__(32)
void conv_qk_kernel(const _Float16* __restrict__ X,
                    const _Float16* __restrict__ WqT, const _Float16* __restrict__ WkT,
                    float* __restrict__ qraw, float* __restrict__ kraw) {
  const int lane = threadIdx.x;
  const int rowm = lane & 15, hi16 = lane >> 4, ks = hi16 * 8;
  const int n0 = blockIdx.x * 32;
  const int m0 = blockIdx.y * 32;
  const int b  = blockIdx.z & 1;
  const _Float16* Wt   = (blockIdx.z >> 1) ? WkT  : WqT;
  float*          outF = (blockIdx.z >> 1) ? kraw : qraw;

  const int p0 = m0 + rowm,      p1 = m0 + 16 + rowm;
  const int t0 = p0 >> 6, h0 = (p0 >> 3) & 7, w0 = p0 & 7;
  const int t1 = p1 >> 6, h1 = (p1 >> 3) & 7, w1 = p1 & 7;

  v8f c00 = {}, c01 = {}, c10 = {}, c11 = {};

  for (int tap = 0; tap < 27; ++tap) {
    const int dt = tap / 9 - 1, dh = (tap / 3) % 3 - 1, dw = tap % 3 - 1;

    const int ti0 = t0 + dt, hi0 = h0 + dh, wi0 = w0 + dw;
    const bool v0 = (ti0 >= 0) & (ti0 < NT) & (hi0 >= 0) & (hi0 < NHs) & (wi0 >= 0) & (wi0 < NWs);
    const _Float16* xr0 = X + ((size_t)b * NP +
        (((v0 ? ti0 : 0) * NHs + (v0 ? hi0 : 0)) * NWs + (v0 ? wi0 : 0))) * NC;

    const int ti1 = t1 + dt, hi1 = h1 + dh, wi1 = w1 + dw;
    const bool v1 = (ti1 >= 0) & (ti1 < NT) & (hi1 >= 0) & (hi1 < NHs) & (wi1 >= 0) & (wi1 < NWs);
    const _Float16* xr1 = X + ((size_t)b * NP +
        (((v1 ? ti1 : 0) * NHs + (v1 ? hi1 : 0)) * NWs + (v1 ? wi1 : 0))) * NC;

    #pragma unroll
    for (int kc = 0; kc < 4; ++kc) {
      v16h a0 = loadAfrag(xr0, v0, kc, ks);
      v16h a1 = loadAfrag(xr1, v1, kc, ks);
      const int ci = kc * 32 + ks;                         // B: lane = N-col, elements = K
      const _Float16* wb = Wt + ((size_t)tap * NC + ci) * NC + n0 + rowm;
      v16h b0, b1;
      #pragma unroll
      for (int j = 0; j < 8; ++j) {
        b0[j] = wb[j * NC]; b0[8 + j] = wb[(16 + j) * NC];
        b1[j] = wb[j * NC + 16]; b1[8 + j] = wb[(16 + j) * NC + 16];
      }
      c00 = __builtin_amdgcn_wmma_f32_16x16x32_f16(false, a0, false, b0, (short)0, c00, false, false);
      c01 = __builtin_amdgcn_wmma_f32_16x16x32_f16(false, a0, false, b1, (short)0, c01, false, false);
      c10 = __builtin_amdgcn_wmma_f32_16x16x32_f16(false, a1, false, b0, (short)0, c10, false, false);
      c11 = __builtin_amdgcn_wmma_f32_16x16x32_f16(false, a1, false, b1, (short)0, c11, false, false);
    }
  }
  // C/D layout: vgpr v -> M = v + 8*hi16, N = rowm
  #pragma unroll
  for (int v = 0; v < 8; ++v) {
    const int r0 = m0 + v + 8 * hi16, r1 = r0 + 16;
    outF[((size_t)b * NP + r0) * NC + n0 + rowm]      = c00[v];
    outF[((size_t)b * NP + r0) * NC + n0 + 16 + rowm] = c01[v];
    outF[((size_t)b * NP + r1) * NC + n0 + rowm]      = c10[v];
    outF[((size_t)b * NP + r1) * NC + n0 + 16 + rowm] = c11[v];
  }
}

// ---------------------------------------------------------------------------
// 4) V projection GEMM via WMMA (K=128), writes f16 [b,h,p,d]
// ---------------------------------------------------------------------------
__global__ __launch_bounds__(32)
void gemm_v_kernel(const _Float16* __restrict__ X, const _Float16* __restrict__ WvT,
                   _Float16* __restrict__ outH) {
  const int lane = threadIdx.x;
  const int rowm = lane & 15, hi16 = lane >> 4, ks = hi16 * 8;
  const int n0 = blockIdx.x * 16;
  const int m0 = blockIdx.y * 16;
  const int b  = blockIdx.z;

  const _Float16* xrow = X + ((size_t)b * NP + (m0 + rowm)) * NC;
  v8f acc = {};
  #pragma unroll
  for (int kc = 0; kc < 4; ++kc) {
    v16h a = loadAfrag(xrow, true, kc, ks);
    const int ci = kc * 32 + ks;
    const _Float16* wb = WvT + (size_t)ci * NC + n0 + rowm;
    v16h bf;
    #pragma unroll
    for (int j = 0; j < 8; ++j) { bf[j] = wb[j * NC]; bf[8 + j] = wb[(16 + j) * NC]; }
    acc = __builtin_amdgcn_wmma_f32_16x16x32_f16(false, a, false, bf, (short)0, acc, false, false);
  }
  #pragma unroll
  for (int v = 0; v < 8; ++v) {
    const int prow = m0 + v + 8 * hi16;
    const int co   = n0 + rowm;
    const int hd = co >> 5, dd = co & 31;
    outH[(((size_t)b * NHEAD + hd) * NP + prow) * ND + dd] = (_Float16)acc[v];
  }
}

// ---------------------------------------------------------------------------
// 5) BN3d training-mode stats: per-channel mean / rsqrt(var+eps) over (B,P)
// ---------------------------------------------------------------------------
__global__ __launch_bounds__(256)
void bn_stats_kernel(const float* __restrict__ qraw, const float* __restrict__ kraw,
                     float* __restrict__ mu, float* __restrict__ inv) {
  const int c = blockIdx.x;
  const float* buf = (blockIdx.y == 0) ? qraw : kraw;
  float s = 0.f, s2 = 0.f;
  for (int r = threadIdx.x; r < NB * NP; r += 256) {
    float v = buf[(size_t)r * NC + c];
    s += v; s2 += v * v;
  }
  __shared__ float sh[256], sh2[256];
  sh[threadIdx.x] = s; sh2[threadIdx.x] = s2;
  __syncthreads();
  for (int off = 128; off > 0; off >>= 1) {
    if (threadIdx.x < off) { sh[threadIdx.x] += sh[threadIdx.x + off];
                             sh2[threadIdx.x] += sh2[threadIdx.x + off]; }
    __syncthreads();
  }
  if (threadIdx.x == 0) {
    const float n = (float)(NB * NP);
    float m  = sh[0] / n;
    float vr = sh2[0] / n - m * m;
    mu [blockIdx.y * NC + c] = m;
    inv[blockIdx.y * NC + c] = rsqrtf(vr + 1e-5f);
  }
}

// ---------------------------------------------------------------------------
// 6) BN apply + head split + f16. q gets 1/gra_sharp folded in.
//    qh: [b,h,p,d]   kth: [b,h,d,p] (transposed for score B-fragments)
// ---------------------------------------------------------------------------
__global__ __launch_bounds__(256)
void bn_apply_kernel(const float* __restrict__ qraw, const float* __restrict__ kraw,
                     const float* __restrict__ mu, const float* __restrict__ inv,
                     const float* __restrict__ gq, const float* __restrict__ bq,
                     const float* __restrict__ gk, const float* __restrict__ bk,
                     const int* __restrict__ gs,
                     _Float16* __restrict__ qh, _Float16* __restrict__ kth) {
  int idx = blockIdx.x * 256 + threadIdx.x;
  if (idx >= NB * NP * NC) return;
  const int c = idx & 127;
  const int p = (idx >> 7) & (NP - 1);
  const int b = idx >> 19;
  const int hd = c >> 5, dd = c & 31;
  if (blockIdx.y == 0) {
    const float inv_gra = 1.0f / (float)gs[0];
    float v = (qraw[idx] - mu[c]) * inv[c] * gq[c] + bq[c];
    qh[(((size_t)b * NHEAD + hd) * NP + p) * ND + dd] = (_Float16)(v * inv_gra);
  } else {
    float v = (kraw[idx] - mu[NC + c]) * inv[NC + c] * gk[c] + bk[c];
    kth[(((size_t)b * NHEAD + hd) * ND + dd) * NP + p] = (_Float16)v;
  }
}

// ---------------------------------------------------------------------------
// 7) flash-attention: one wave -> 32 queries (2 q-tiles), K/V frags reused 2x.
//    Each lane owns one score row: max + exp-in-place-in-LDS (1 exp/element).
// ---------------------------------------------------------------------------
__global__ __launch_bounds__(32)
void attn_kernel(const _Float16* __restrict__ qh, const _Float16* __restrict__ kth,
                 const _Float16* __restrict__ vh, float* __restrict__ out) {
  __shared__ float Sbuf[32][33];
  __shared__ float aBuf[32], lBuf[32];

  const int lane = threadIdx.x;
  const int rowm = lane & 15, hi16 = lane >> 4, ks = hi16 * 8;
  const int q0 = blockIdx.x * 32;
  const int h  = blockIdx.y;
  const int b  = blockIdx.z;
  const size_t bh = (size_t)b * NHEAD + h;

  // two q A-fragments (1/gra_sharp already folded in)
  v16h qa0, qa1;
  {
    const _Float16* qr0 = qh + (bh * NP + (q0 + rowm)) * ND;
    const _Float16* qr1 = qr0 + 16 * ND;
    v8h lo0 = *(const v8h*)(qr0 + ks), hv0 = *(const v8h*)(qr0 + ks + 16);
    v8h lo1 = *(const v8h*)(qr1 + ks), hv1 = *(const v8h*)(qr1 + ks + 16);
    #pragma unroll
    for (int j = 0; j < 8; ++j) {
      qa0[j] = lo0[j]; qa0[8 + j] = hv0[j];
      qa1[j] = lo1[j]; qa1[8 + j] = hv1[j];
    }
  }

  float m_run = -3.0e38f, l_run = 0.0f;     // stats for row = lane (0..31)
  v8f O00 = {}, O01 = {}, O10 = {}, O11 = {};

  for (int k0 = 0; k0 < NP; k0 += 32) {
    // scores: B-fragment lane = key column, elements = d
    const _Float16* krow = kth + (bh * ND + ks) * NP + k0 + rowm;
    v16h kb0, kb1;
    #pragma unroll
    for (int j = 0; j < 8; ++j) {
      kb0[j] = krow[j * NP]; kb0[8 + j] = krow[(16 + j) * NP];
      kb1[j] = krow[j * NP + 16]; kb1[8 + j] = krow[(16 + j) * NP + 16];
    }
    v8f s00 = {}, s01 = {}, s10 = {}, s11 = {};
    s00 = __builtin_amdgcn_wmma_f32_16x16x32_f16(false, qa0, false, kb0, (short)0, s00, false, false);
    s01 = __builtin_amdgcn_wmma_f32_16x16x32_f16(false, qa0, false, kb1, (short)0, s01, false, false);
    s10 = __builtin_amdgcn_wmma_f32_16x16x32_f16(false, qa1, false, kb0, (short)0, s10, false, false);
    s11 = __builtin_amdgcn_wmma_f32_16x16x32_f16(false, qa1, false, kb1, (short)0, s11, false, false);

    // D layout -> LDS [query 0..31][key 0..31]
    #pragma unroll
    for (int v = 0; v < 8; ++v) {
      const int r = v + 8 * hi16;
      Sbuf[r][rowm]           = s00[v];
      Sbuf[r][rowm + 16]      = s01[v];
      Sbuf[16 + r][rowm]      = s10[v];
      Sbuf[16 + r][rowm + 16] = s11[v];
    }
    __syncthreads();

    // online softmax; lane owns row = lane; exp in place (one exp per element)
    float tmax = -3.0e38f;
    #pragma unroll
    for (int j = 0; j < 32; ++j) tmax = fmaxf(tmax, Sbuf[lane][j]);
    const float m_new = fmaxf(m_run, tmax);
    const float alpha = __expf(m_run - m_new);
    float ps = 0.f;
    #pragma unroll
    for (int j = 0; j < 32; ++j) {
      float e = __expf(Sbuf[lane][j] - m_new);
      Sbuf[lane][j] = e;
      ps += e;
    }
    l_run = alpha * l_run + ps;
    m_run = m_new;
    aBuf[lane] = alpha;
    __syncthreads();

    // build P A-fragments from exp'd LDS
    v16h pa0, pa1;
    #pragma unroll
    for (int j = 0; j < 8; ++j) {
      pa0[j]     = (_Float16)Sbuf[rowm][ks + j];
      pa0[8 + j] = (_Float16)Sbuf[rowm][ks + 16 + j];
      pa1[j]     = (_Float16)Sbuf[16 + rowm][ks + j];
      pa1[8 + j] = (_Float16)Sbuf[16 + rowm][ks + 16 + j];
    }
    // rescale accumulators (C-fragment row mapping)
    #pragma unroll
    for (int v = 0; v < 8; ++v) {
      const float al0 = aBuf[v + 8 * hi16];
      const float al1 = aBuf[16 + v + 8 * hi16];
      O00[v] *= al0; O01[v] *= al0;
      O10[v] *= al1; O11[v] *= al1;
    }

    // O += P @ V : B-fragment lane = d column, elements = key
    const _Float16* vrow = vh + (bh * NP + (k0 + ks)) * ND + rowm;
    v16h vb0, vb1;
    #pragma unroll
    for (int j = 0; j < 8; ++j) {
      vb0[j] = vrow[j * ND]; vb0[8 + j] = vrow[(16 + j) * ND];
      vb1[j] = vrow[j * ND + 16]; vb1[8 + j] = vrow[(16 + j) * ND + 16];
    }
    O00 = __builtin_amdgcn_wmma_f32_16x16x32_f16(false, pa0, false, vb0, (short)0, O00, false, false);
    O01 = __builtin_amdgcn_wmma_f32_16x16x32_f16(false, pa0, false, vb1, (short)0, O01, false, false);
    O10 = __builtin_amdgcn_wmma_f32_16x16x32_f16(false, pa1, false, vb0, (short)0, O10, false, false);
    O11 = __builtin_amdgcn_wmma_f32_16x16x32_f16(false, pa1, false, vb1, (short)0, O11, false, false);
    __syncthreads();       // Sbuf reused next iter
  }

  lBuf[lane] = l_run;
  __syncthreads();
  #pragma unroll
  for (int v = 0; v < 8; ++v) {
    const float li0 = 1.0f / lBuf[v + 8 * hi16];
    const float li1 = 1.0f / lBuf[16 + v + 8 * hi16];
    const int r0 = q0 + v + 8 * hi16, r1 = r0 + 16;
    out[((size_t)b * NP + r0) * NC + h * ND + rowm]      = O00[v] * li0;
    out[((size_t)b * NP + r0) * NC + h * ND + 16 + rowm] = O01[v] * li0;
    out[((size_t)b * NP + r1) * NC + h * ND + rowm]      = O10[v] * li1;
    out[((size_t)b * NP + r1) * NC + h * ND + 16 + rowm] = O11[v] * li1;
  }
}

// ---------------------------------------------------------------------------
extern "C" void kernel_launch(void* const* d_in, const int* in_sizes, int n_in,
                              void* d_out, int out_size, void* d_ws, size_t ws_size,
                              hipStream_t stream) {
  const float* x  = (const float*)d_in[0];
  const float* Wq = (const float*)d_in[1];
  const float* gq = (const float*)d_in[2];
  const float* bq = (const float*)d_in[3];
  const float* Wk = (const float*)d_in[4];
  const float* gk = (const float*)d_in[5];
  const float* bk = (const float*)d_in[6];
  const float* Wv = (const float*)d_in[7];
  const int*   gs = (const int*)  d_in[8];
  float* out = (float*)d_out;

  // workspace partition (256B aligned)
  char* ws = (char*)d_ws;
  size_t off = 0;
  auto alloc = [&](size_t bytes) { char* p = ws + off; off = (off + bytes + 255) & ~(size_t)255; return p; };
  _Float16* xh   = (_Float16*)alloc((size_t)NB * NP * NC * 2);
  _Float16* WqT  = (_Float16*)alloc((size_t)27 * NC * NC * 2);
  _Float16* WkT  = (_Float16*)alloc((size_t)27 * NC * NC * 2);
  _Float16* WvT  = (_Float16*)alloc((size_t)NC * NC * 2);
  float*    qraw = (float*)   alloc((size_t)NB * NP * NC * 4);
  float*    kraw = (float*)   alloc((size_t)NB * NP * NC * 4);
  float*    mu   = (float*)   alloc(2 * NC * 4);
  float*    inv  = (float*)   alloc(2 * NC * 4);
  _Float16* qhB  = (_Float16*)alloc((size_t)NB * NP * NC * 2);
  _Float16* kthB = (_Float16*)alloc((size_t)NB * NP * NC * 2);
  _Float16* vhB  = (_Float16*)alloc((size_t)NB * NP * NC * 2);
  (void)ws_size; (void)n_in; (void)in_sizes; (void)out_size;

  const int nelem = NB * NP * NC;
  xcvt_kernel<<<dim3((nelem + 255) / 256), dim3(256), 0, stream>>>(x, xh, nelem);
  weight_prep_kernel<<<dim3(64, 3), dim3(256), 0, stream>>>(Wq, Wk, Wv, WqT, WkT, WvT);

  // Q and K CDC-convs in one launch: grid.z = {b | qk<<1}
  conv_qk_kernel<<<dim3(NC / 32, NP / 32, 4), dim3(32), 0, stream>>>(xh, WqT, WkT, qraw, kraw);
  gemm_v_kernel<<<dim3(NC / 16, NP / 16, NB), dim3(32), 0, stream>>>(xh, WvT, vhB);

  bn_stats_kernel<<<dim3(NC, 2), dim3(256), 0, stream>>>(qraw, kraw, mu, inv);
  bn_apply_kernel<<<dim3((nelem + 255) / 256, 2), dim3(256), 0, stream>>>(
      qraw, kraw, mu, inv, gq, bq, gk, bk, gs, qhB, kthB);

  attn_kernel<<<dim3(NP / 32, NHEAD, NB), dim3(32), 0, stream>>>(qhB, kthB, vhB, out);
}
